// BrainInspiredRNN_54571854463280
// MI455X (gfx1250) — matrix operand order repaired and checked by reading, not compiled, not run on hardware
//
#include <hip/hip_runtime.h>
#include <math.h>

typedef __attribute__((ext_vector_type(16))) _Float16 v16h;
typedef __attribute__((ext_vector_type(8)))  float    v8f;
typedef __attribute__((ext_vector_type(4)))  float    v4f;

#define RANK 6

static __device__ __forceinline__ v8f wmma16(v16h a, v16h b, v8f c) {
  // D = A(16x32 f16) * B(32x16 f16) + C(16x16 f32)
  return __builtin_amdgcn_wmma_f32_16x16x32_f16(false, a, false, b, (short)0, c, false, false);
}

static __device__ __forceinline__ float fast_tanh(float x) {
#if __has_builtin(__builtin_amdgcn_tanhf)
  return __builtin_amdgcn_tanhf(x);            // native v_tanh_f32 (1 TRANS op)
#else
  x = fminf(10.0f, fmaxf(-10.0f, x));
  float t = __builtin_amdgcn_exp2f(x * 2.8853900817779268f);  // e^(2x)
  return (t - 1.0f) * __builtin_amdgcn_rcpf(t + 1.0f);
#endif
}

// Pure-VALU half-wave swap (shfl_xor 16 in wave32): v_permlanex16_b32 with
// identity lane selects reads the same lane index from the opposite 16-lane row.
static __device__ __forceinline__ float xor16(float x) {
  int xi = __builtin_bit_cast(int, x);
  int r  = __builtin_amdgcn_permlanex16(xi, xi, 0x76543210u, 0xFEDCBA98u,
                                        false, false);
  return __builtin_bit_cast(float, r);
}

static __device__ __forceinline__ void cvt4(v16h& d, int base, v4f v) {
  d[base + 0] = (_Float16)v[0];
  d[base + 1] = (_Float16)v[1];
  d[base + 2] = (_Float16)v[2];
  d[base + 3] = (_Float16)v[3];
}

// One wave (32 lanes) per 16-row batch tile. Transposed orientation:
//   pre^T[32n x 16b] = W_rec[32x32] * H^T[32x16]  (2 WMMA; 1-WMMA-deep chain)
//   xp^T  = W_in * X^T   (4 WMMA, off critical path)
//   y^T   = W_out * H^T  (2 WMMA, off critical path)
__global__ __launch_bounds__(32) void rnn_fused_scan(
    const float* __restrict__ inputs,   // [B,T,64]
    const float* __restrict__ W_in,     // [32,64]
    const float* __restrict__ mvec,     // [32]
    const float* __restrict__ nvec,     // [32]
    const float* __restrict__ Mmat,     // [32,6]
    const float* __restrict__ Nm,       // [32,6]
    const float* __restrict__ bias,     // [32]
    const float* __restrict__ W_out,    // [32,32]
    const float* __restrict__ b_out,    // [32]
    float* __restrict__ out,            // [B,T,32] then [B,32]
    int B, int T)
{
  __shared__ float yscr[16 * 36];       // 36-float stride: 16B-aligned rows
  const int lane  = threadIdx.x & 31;
  const int bTile = blockIdx.x;
  const int lm = lane & 15;
  const int hi = lane >> 4;             // half-wave id

  // ---- Build f16 A-operands once (lane = M row; 16 halves = K slice) ----
  // A k order used here (contiguous, not ISA table order): lanes<16 hold K {0..15},
  // lanes>=16 hold K {16..31}; half j -> K = j + 16*hi (matches B operands).
  v16h WinA[2][2], WrecA[2], WoutA[2];
  #pragma unroll
  for (int nb = 0; nb < 2; ++nb) {
    const int row = nb * 16 + lm;
    const float mrow = mvec[row];
    float Mr[RANK];
    #pragma unroll
    for (int r = 0; r < RANK; ++r) Mr[r] = Mmat[row * RANK + r];
    #pragma unroll
    for (int j = 0; j < 16; ++j) {
      const int k = j + (hi << 4);
      WinA[nb][0][j] = (_Float16)W_in[row * 64 + k];
      WinA[nb][1][j] = (_Float16)W_in[row * 64 + 32 + k];
      // W_rec[row][k] = m[row]*n[k] + sum_r M[row,r]*N[k,r]
      float w = mrow * nvec[k];
      #pragma unroll
      for (int r = 0; r < RANK; ++r) w += Mr[r] * Nm[k * RANK + r];
      WrecA[nb][j] = (_Float16)w;
      WoutA[nb][j] = (_Float16)W_out[row * 32 + k];
    }
  }

  // C-layout bias tiles: VGPR i, lane -> neuron = i + 8*hi (+16 for tile1)
  v8f biasC[2], boutC[2];
  #pragma unroll
  for (int i = 0; i < 8; ++i) {
    biasC[0][i] = bias[i + 8 * hi];
    biasC[1][i] = bias[16 + i + 8 * hi];
    boutC[0][i] = b_out[i + 8 * hi];
    boutC[1][i] = b_out[16 + i + 8 * hi];
  }

  // f32 state in C/D layout (neuron in VGPR slot, batch in lane)
  v8f S0 = {0,0,0,0,0,0,0,0}, S1 = {0,0,0,0,0,0,0,0};
  // f16 state in B-operand layout (lane = batch col, halves = neuron K)
  v16h Hb;
  #pragma unroll
  for (int j = 0; j < 16; ++j) Hb[j] = (_Float16)0.0f;

  // B-operand X^T: lane b holds K dims (hi*16 .. hi*16+15), contiguous floats
  const float* xbase =
      inputs + ((size_t)(bTile * 16 + lm) * (size_t)T) * 64 + hi * 16;
  const float kLeak = 0.9f, kAlpha = 0.1f;

  // Per-lane persistent output pointers: 4 b128 stores/step cover 16 rows x 32 f32
  float* optr[4];
  int ldsOff[4];
  #pragma unroll
  for (int s = 0; s < 4; ++s) {
    const int id    = s * 32 + lane;
    const int row   = id >> 3;     // 0..15 (batch row in tile)
    const int chunk = id & 7;      // 0..7  (4-float chunk of the 32-float row)
    optr[s]   = out + ((size_t)(bTile * 16 + row) * (size_t)T) * 32 + chunk * 4;
    ldsOff[s] = row * 36 + chunk * 4;
  }

  v4f xn[8];                              // staged X(t), double-buffered

  auto loadX = [&](const float* xq) {     // streamed once -> non-temporal
    #pragma unroll
    for (int q = 0; q < 8; ++q) {
      const int off = 4 * q + ((q >= 4) ? 16 : 0);   // 0..15 then 32..47
      xn[q] = __builtin_nontemporal_load((const v4f*)(xq + off));
    }
  };
  auto makeXb = [&](v16h& Xb0, v16h& Xb1) {
    #pragma unroll
    for (int q = 0; q < 4; ++q) {
      cvt4(Xb0, 4 * q, xn[q]);
      cvt4(Xb1, 4 * q, xn[4 + q]);
    }
  };

  auto doStep = [&](v16h Xb0, v16h Xb1) {
    // xp^T + bias  (off critical path)
    v8f c0 = wmma16(WinA[0][0], Xb0, biasC[0]);
    c0     = wmma16(WinA[0][1], Xb1, c0);
    v8f c1 = wmma16(WinA[1][0], Xb0, biasC[1]);
    c1     = wmma16(WinA[1][1], Xb1, c1);

    // pre^T = W_rec * h_t^T + xp^T + bias   (critical: 1 WMMA deep per tile)
    v8f p0 = wmma16(WrecA[0], Hb, c0);
    v8f p1 = wmma16(WrecA[1], Hb, c1);

    // leaky tanh update (native TRANS tanh keeps the chain short)
    #pragma unroll
    for (int i = 0; i < 8; ++i) {
      S0[i] = kLeak * S0[i] + kAlpha * fast_tanh(p0[i]);
      S1[i] = kLeak * S1[i] + kAlpha * fast_tanh(p1[i]);
    }

    // Repack D-layout f32 state -> B-operand f16: pure-VALU half swap + select
    #pragma unroll
    for (int i = 0; i < 8; ++i) {
      float a0 = xor16(S0[i]);                  // partner half's neurons
      float a1 = xor16(S1[i]);
      Hb[i]     = (_Float16)(hi ? a1 : S0[i]);  // K = i      (+16 if hi)
      Hb[8 + i] = (_Float16)(hi ? S1[i] : a0);  // K = 8 + i  (+16 if hi)
    }

    // y^T = W_out * h_{t+1}^T + b_out   (off critical path)
    v8f y0 = wmma16(WoutA[0], Hb, boutC[0]);
    v8f y1 = wmma16(WoutA[1], Hb, boutC[1]);

    // Stage tiles into LDS (4x ds_store_b128), transposing o<->b
    const int sb = lm * 36 + 8 * hi;
    {
      v4f w0 = {y0[0], y0[1], y0[2], y0[3]};
      v4f w1 = {y0[4], y0[5], y0[6], y0[7]};
      v4f w2 = {y1[0], y1[1], y1[2], y1[3]};
      v4f w3 = {y1[4], y1[5], y1[6], y1[7]};
      *(v4f*)&yscr[sb]      = w0;   // (batch lm, o = 8*hi + 0..3)
      *(v4f*)&yscr[sb + 4]  = w1;   // o = 8*hi + 4..7
      *(v4f*)&yscr[sb + 16] = w2;   // o = 16 + 8*hi + 0..3
      *(v4f*)&yscr[sb + 20] = w3;   // o = 16 + 8*hi + 4..7
    }
    // Drain: 4x ds_load_b128 + 4x coalesced non-temporal global_store_b128.
    // Same-wave DS ops are in-order, so loads see this step's stores and
    // complete before next step's stores overwrite the buffer.
    #pragma unroll
    for (int s = 0; s < 4; ++s) {
      v4f v = *(const v4f*)&yscr[ldsOff[s]];
      __builtin_nontemporal_store(v, (v4f*)optr[s]);
      optr[s] += 32;                  // advance one timestep (128 B)
    }
  };

  // ---- main scan: peeled last iteration keeps the loop clamp-free ----
  loadX(xbase);                         // X(0)
  const float* xnext = xbase + 64;
  for (int t = 0; t < T - 1; ++t) {
    v16h Xb0, Xb1;
    makeXb(Xb0, Xb1);                   // consume X(t)
    __builtin_prefetch(xnext + 64, 0, 0);  // global_prefetch_b8, two steps ahead
    loadX(xnext);                       // X(t+1): lands under WMMA/tanh/stores
    xnext += 64;
    doStep(Xb0, Xb1);
  }
  {
    v16h Xb0, Xb1;
    makeXb(Xb0, Xb1);                   // consume X(T-1), nothing left to load
    doStep(Xb0, Xb1);
  }

  // h_final appended after outputs (b128 stores)
  const size_t hoff = (size_t)B * (size_t)T * 32;
  float* hrow = out + hoff + (size_t)(bTile * 16 + lm) * 32 + 8 * hi;
  {
    v4f h0 = {S0[0], S0[1], S0[2], S0[3]};
    v4f h1 = {S0[4], S0[5], S0[6], S0[7]};
    v4f h2 = {S1[0], S1[1], S1[2], S1[3]};
    v4f h3 = {S1[4], S1[5], S1[6], S1[7]};
    *(v4f*)(hrow)      = h0;
    *(v4f*)(hrow + 4)  = h1;
    *(v4f*)(hrow + 16) = h2;
    *(v4f*)(hrow + 20) = h3;
  }
}

extern "C" void kernel_launch(void* const* d_in, const int* in_sizes, int n_in,
                              void* d_out, int out_size, void* d_ws, size_t ws_size,
                              hipStream_t stream) {
  const float* inputs = (const float*)d_in[0];
  const float* W_in   = (const float*)d_in[1];
  const float* m      = (const float*)d_in[2];
  const float* n      = (const float*)d_in[3];
  const float* M      = (const float*)d_in[4];
  const float* Nm     = (const float*)d_in[5];
  const float* bias   = (const float*)d_in[6];
  const float* W_out  = (const float*)d_in[7];
  const float* b_out  = (const float*)d_in[8];
  float* out = (float*)d_out;

  const long long in0 = in_sizes[0];                 // B*T*64
  const long long BT  = in0 / 64;
  const long long B   = ((long long)out_size - BT * 32) / 32;  // out = 32*B*T + 32*B
  const long long T   = BT / B;

  dim3 grid((unsigned)(B / 16)), block(32);
  rnn_fused_scan<<<grid, block, 0, stream>>>(inputs, W_in, m, n, M, Nm, bias,
                                             W_out, b_out, out, (int)B, (int)T);
}
